// ProbAttention_9225589752334
// MI455X (gfx1250) — matrix operand order repaired and checked by reading, not compiled, not run on hardware
//
#include <hip/hip_runtime.h>
#include <hip/hip_bf16.h>
#include <stdint.h>

typedef _Float16 v16h __attribute__((ext_vector_type(16)));
typedef _Float16 h2   __attribute__((ext_vector_type(2)));
typedef float    v8f  __attribute__((ext_vector_type(8)));

#define B_ 32
#define L_ 2048
#define S_ 2048
#define E_ 64
#define D_ 64

#define BM 64          // query rows per block (4 waves x 16 rows)
#define BN 64          // keys per S-tile iteration
#define NWAVES 4
#define STR 72         // LDS row stride in halves: 144B = 9*16B (b128-aligned, conflict-benign)

#if defined(__has_builtin)
#if __has_builtin(__builtin_amdgcn_global_load_async_to_lds_b128)
#define HAVE_ASYNC_LDS 1
#endif
#endif

// the builtin wants pointers to 16-byte int vectors (per hipcc diagnostic)
typedef __attribute__((__vector_size__(4 * sizeof(int)))) int i128v;
typedef __attribute__((address_space(1))) i128v gi128;
typedef __attribute__((address_space(3))) i128v li128;

// copy 16 bytes global(f16) -> LDS, async DMA if available
__device__ __forceinline__ void copy16_g2l(const _Float16* g, _Float16* l) {
#ifdef HAVE_ASYNC_LDS
    __builtin_amdgcn_global_load_async_to_lds_b128(
        (gi128*)(uintptr_t)g, (li128*)(uint32_t)(uintptr_t)l, 0, 0);
#else
    const uint4 d = *(const uint4*)g;
    uint32_t* lp = (uint32_t*)l;
    lp[0] = d.x; lp[1] = d.y; lp[2] = d.z; lp[3] = d.w;
#endif
}

__device__ __forceinline__ void wait_g2l() {
#ifdef HAVE_ASYNC_LDS
#if __has_builtin(__builtin_amdgcn_s_wait_asynccnt)
    __builtin_amdgcn_s_wait_asynccnt(0);
#else
    asm volatile("s_wait_asynccnt 0x0" ::: "memory");
#endif
#endif
}

// kbase for the 16x32 f16 A-fragment: VGPR v, lane-half g
__device__ __forceinline__ int a_kbase(int v, int g) {
    return ((v >> 2) << 4) + ((v & 3) << 1) + (g << 3);
}

// ---- pre-pass: fp32 -> f16 conversions into workspace ----
__global__ __launch_bounds__(256) void cvt_k_f16(const float* __restrict__ K,
                                                 _Float16* __restrict__ dst) {
    const size_t i = (size_t)blockIdx.x * 256 + threadIdx.x;   // [b][s][e], same layout
    dst[i] = (_Float16)K[i];
}

__global__ __launch_bounds__(256) void cvt_vt_f16(const float* __restrict__ V,
                                                  _Float16* __restrict__ dst) {
    // dst layout: [b][tile][d][key], 64x64 f16 per tile (contiguous 8KB)
    const size_t i = (size_t)blockIdx.x * 256 + threadIdx.x;
    const int key = (int)(i & 63);
    const int d   = (int)((i >> 6) & 63);
    const size_t tb = i >> 12;                                  // b*32 + tile
    dst[i] = (_Float16)V[(tb * 64 + key) * 64 + d];
}

template <bool USEWS>
__global__ __launch_bounds__(128, 1)
void flash_attn_wmma_f16(const float* __restrict__ Q,
                         const float* __restrict__ K,
                         const float* __restrict__ V,
                         const _Float16* __restrict__ wsK,
                         const _Float16* __restrict__ wsV,
                         float* __restrict__ O) {
    __shared__ _Float16 k_lds[BN * STR];            // K tile, [key][e]
    __shared__ _Float16 vt_lds[D_ * STR];           // V tile transposed, [d][key]
    __shared__ _Float16 p_lds[NWAVES][16 * STR];    // per-wave P tile, [row][key]

    const int tid  = threadIdx.x;
    const int wave = tid >> 5;
    const int lane = tid & 31;
    const int g    = lane >> 4;      // 16-lane half
    const int ln   = lane & 15;

    const int b  = blockIdx.y;
    const int m0 = blockIdx.x * BM + wave * 16;      // this wave's first query row

    // fold 1/sqrt(E) and log2(e) into Q so softmax is exp2(s - m)
    const float cscale = 0.125f * 1.44269504088896340736f;

    // ---- load Q A-fragments once (2 chunks of E=32), fp32 -> f16, pre-scaled ----
    v16h qa[2];
    const float* qrow = Q + ((size_t)b * L_ + (size_t)(m0 + ln)) * E_;
#pragma unroll
    for (int f = 0; f < 2; ++f) {
#pragma unroll
        for (int v = 0; v < 8; ++v) {
            const int e = f * 32 + a_kbase(v, g);
            qa[f][2 * v + 0] = (_Float16)(qrow[e + 0] * cscale);
            qa[f][2 * v + 1] = (_Float16)(qrow[e + 1] * cscale);
        }
    }

    // constant all-ones B fragment: row-sum of P via WMMA (P x ones)
    v16h onesb;
#pragma unroll
    for (int i = 0; i < 16; ++i) onesb[i] = (_Float16)1.0f;

    // ---- running softmax state + O accumulators (C layout: row = v + 8*g) ----
    v8f   o_acc[4];
    float m_run[8], l_run[8];
#pragma unroll
    for (int v = 0; v < 8; ++v) { m_run[v] = -__builtin_inff(); l_run[v] = 0.0f; }
#pragma unroll
    for (int dt = 0; dt < 4; ++dt)
#pragma unroll
        for (int v = 0; v < 8; ++v) o_acc[dt][v] = 0.0f;

    for (int j0 = 0; j0 < S_; j0 += BN) {
        if constexpr (USEWS) {
            // ---- stage pre-converted f16 K / V^T tiles via (async) 16B copies ----
            const _Float16* kt = wsK + ((size_t)b * S_ + j0) * E_;
            const _Float16* vt = wsV + ((size_t)b * (S_ / BN) + (j0 / BN)) * (size_t)(D_ * BN);
#pragma unroll
            for (int i = 0; i < 4; ++i) {
                const int c    = tid + i * 128;      // 16B chunk id, 0..511
                const int row  = c >> 3;             // key (K) / d (V)
                const int colh = (c & 7) * 8;        // halves within the 128B row
                copy16_g2l(kt + (size_t)c * 8, &k_lds[row * STR + colh]);
                copy16_g2l(vt + (size_t)c * 8, &vt_lds[row * STR + colh]);
            }
            wait_g2l();
        } else {
            // ---- fallback: cooperative f32 load + convert (no workspace) ----
            const float* kb = K + ((size_t)b * S_ + j0) * E_;
#pragma unroll
            for (int i = 0; i < 16; ++i) {
                const int p   = tid + i * 128;       // pair index, 0..2047
                const int key = p >> 5;
                const int e   = (p & 31) << 1;
                const float2 f2 = *(const float2*)(kb + (size_t)key * E_ + e);
                h2 hh; hh.x = (_Float16)f2.x; hh.y = (_Float16)f2.y;
                *(h2*)&k_lds[key * STR + e] = hh;
            }
            const float* vb = V + ((size_t)b * S_ + j0) * D_;
#pragma unroll
            for (int i = 0; i < 16; ++i) {
                const int p   = tid + i * 128;
                const int key = p >> 5;
                const int d   = (p & 31) << 1;
                const float2 f2 = *(const float2*)(vb + (size_t)key * D_ + d);
                vt_lds[(d + 0) * STR + key] = (_Float16)f2.x;
                vt_lds[(d + 1) * STR + key] = (_Float16)f2.y;
            }
        }
        __syncthreads();

        // ---- scores S = Q * K^T : 4 n-tiles x 2 E-chunks = 8 WMMAs ----
        v8f s[4];
#pragma unroll
        for (int nt = 0; nt < 4; ++nt) {
            v8f acc;
#pragma unroll
            for (int v = 0; v < 8; ++v) acc[v] = 0.0f;
#pragma unroll
            for (int f = 0; f < 2; ++f) {
                v16h kbf;   // B fragment 32x16: K(e) = a_kbase(v, g) + h, N = ln
#pragma unroll
                for (int v = 0; v < 8; ++v) {
                    const int e = f * 32 + a_kbase(v, g);
                    h2 hh = *(const h2*)&k_lds[(ln + nt * 16) * STR + e];
                    kbf[2 * v] = hh.x; kbf[2 * v + 1] = hh.y;
                }
                acc = __builtin_amdgcn_wmma_f32_16x16x32_f16(
                          false, qa[f], false, kbf, (short)0, acc, false, false);
            }
            s[nt] = acc;
        }

        // ---- online softmax: row max across 64 keys (local + shfl_xor tree) ----
        float m_new[8], alpha[8];
#pragma unroll
        for (int v = 0; v < 8; ++v) {
            float mx = fmaxf(fmaxf(s[0][v], s[1][v]), fmaxf(s[2][v], s[3][v]));
#pragma unroll
            for (int off = 1; off < 16; off <<= 1)
                mx = fmaxf(mx, __shfl_xor(mx, off, 32));
            m_new[v] = fmaxf(m_run[v], mx);
            alpha[v] = exp2f(m_run[v] - m_new[v]);
            m_run[v] = m_new[v];
        }

        // ---- P = exp2(S - m), spill P (f16) to LDS ----
#pragma unroll
        for (int nt = 0; nt < 4; ++nt) {
#pragma unroll
            for (int v = 0; v < 8; ++v) {
                const float p = exp2f(s[nt][v] - m_new[v]);
                p_lds[wave][(v + 8 * g) * STR + (ln + nt * 16)] = (_Float16)p;
            }
        }

        // ---- rescale O accumulators by alpha ----
#pragma unroll
        for (int dt = 0; dt < 4; ++dt)
#pragma unroll
            for (int v = 0; v < 8; ++v) o_acc[dt][v] *= alpha[v];

        // ---- reload P as A fragments (same-wave LDS, 2 key-chunks) ----
        v16h pa[2];
#pragma unroll
        for (int kc = 0; kc < 2; ++kc) {
#pragma unroll
            for (int v = 0; v < 8; ++v) {
                const int kk = kc * 32 + a_kbase(v, g);
                h2 hh = *(const h2*)&p_lds[wave][ln * STR + kk];
                pa[kc][2 * v] = hh.x; pa[kc][2 * v + 1] = hh.y;
            }
        }

        // ---- row sums via WMMA: sacc = P x ones (C layout matches l_run rows) ----
        v8f sacc;
#pragma unroll
        for (int v = 0; v < 8; ++v) sacc[v] = 0.0f;
#pragma unroll
        for (int kc = 0; kc < 2; ++kc)
            sacc = __builtin_amdgcn_wmma_f32_16x16x32_f16(
                       false, pa[kc], false, onesb, (short)0, sacc, false, false);
#pragma unroll
        for (int v = 0; v < 8; ++v)
            l_run[v] = l_run[v] * alpha[v] + sacc[v];

        // ---- O += P * V : 4 d-tiles x 2 key-chunks = 8 WMMAs ----
#pragma unroll
        for (int dt = 0; dt < 4; ++dt) {
#pragma unroll
            for (int kc = 0; kc < 2; ++kc) {
                v16h vbf;   // B fragment 32x16: K(key) = kc*32 + a_kbase(v, g) + h, N(d) = ln
#pragma unroll
                for (int v = 0; v < 8; ++v) {
                    const int key = kc * 32 + a_kbase(v, g);
                    h2 hh = *(const h2*)&vt_lds[(ln + dt * 16) * STR + key];
                    vbf[2 * v] = hh.x; vbf[2 * v + 1] = hh.y;
                }
                o_acc[dt] = __builtin_amdgcn_wmma_f32_16x16x32_f16(
                                false, pa[kc], false, vbf, (short)0, o_acc[dt], false, false);
            }
        }
        __syncthreads();
    }

    // ---- normalize and store fp32 output ----
#pragma unroll
    for (int v = 0; v < 8; ++v) {
        const float inv = 1.0f / l_run[v];
        const int row = m0 + v + 8 * g;
        float* orow = O + ((size_t)b * L_ + row) * D_;
#pragma unroll
        for (int dt = 0; dt < 4; ++dt)
            orow[ln + dt * 16] = o_acc[dt][v] * inv;
    }
}

extern "C" void kernel_launch(void* const* d_in, const int* in_sizes, int n_in,
                              void* d_out, int out_size, void* d_ws, size_t ws_size,
                              hipStream_t stream) {
    const float* Q = (const float*)d_in[0];
    const float* K = (const float*)d_in[1];
    const float* V = (const float*)d_in[2];
    float* O = (float*)d_out;

    constexpr size_t NKV     = (size_t)B_ * S_ * E_;       // elements per tensor
    constexpr size_t WS_NEED = NKV * 2 * sizeof(_Float16); // f16 K + f16 V^T

    dim3 grid(L_ / BM, B_);   // (32, 32)

    if (ws_size >= WS_NEED) {
        _Float16* wsK = (_Float16*)d_ws;
        _Float16* wsV = wsK + NKV;
        cvt_k_f16 <<<(unsigned)(NKV / 256), 256, 0, stream>>>(K, wsK);
        cvt_vt_f16<<<(unsigned)(NKV / 256), 256, 0, stream>>>(V, wsV);
        flash_attn_wmma_f16<true><<<grid, 128, 0, stream>>>(Q, K, V, wsK, wsV, O);
    } else {
        flash_attn_wmma_f16<false><<<grid, 128, 0, stream>>>(Q, K, V, nullptr, nullptr, O);
    }
}
